// DeformableConv2d_35012573397376
// MI455X (gfx1250) — matrix operand order repaired and checked by reading, not compiled, not run on hardware
//
#include <hip/hip_runtime.h>

// ---------------------------------------------------------------------------
// Deformable conv forward for MI455X (gfx1250, wave32, WMMA).
// Both GEMMs (offset conv + regression) run on v_wmma_f32_16x16x32_f16 with
// fp32 accumulation; bilinear sampling is fp32 VALU feeding f16 LDS tiles.
// Contraction index permuted to k = tap*64 + c so each K=32 WMMA step is a
// fixed tap with contiguous channels. The main loop double-buffers the
// sampled tile so tap t+1's gathers overlap tap t's WMMA stream (one barrier
// per tap).
// ---------------------------------------------------------------------------

typedef __attribute__((ext_vector_type(16))) _Float16 v16h;
typedef __attribute__((ext_vector_type(8)))  _Float16 v8h;
typedef __attribute__((ext_vector_type(8)))  float    v8f;

#define B_    8
#define C_    64
#define H_    128
#define W_    128
#define HW_   (H_ * W_)
#define KTOT  576      // C_ * 9
#define OC_   64
#define OFFC  18
#define XTILE 64       // pixels per workgroup (one row segment)

// workspace layout (bytes)
#define WS_OFFS_BYTES ((size_t)B_ * OFFC * HW_ * 4)          // fp32 offsets, NCHW
#define WS_BW_OFF     WS_OFFS_BYTES                          // f16 Bw[576][64]
#define WS_BO_OFF     (WS_BW_OFF + (size_t)KTOT * OC_ * 2)   // f16 Bo[576][32]

// LDS row pitch for A-side tiles (f16 elements) — 144B to spread banks
#define A_LD 72

__device__ __forceinline__ v8f wmma_f16(v16h a, v16h b, v8f c) {
    return __builtin_amdgcn_wmma_f32_16x16x32_f16(false, a, false, b,
                                                  (short)0, c, false, false);
}

__device__ __forceinline__ v16h cat8(v8h lo, v8h hi) {
    return __builtin_shufflevector(lo, hi, 0, 1, 2, 3, 4, 5, 6, 7,
                                            8, 9, 10, 11, 12, 13, 14, 15);
}

// ---------------------------------------------------------------------------
// Kernel 0: permute + f32->f16 convert weights into ws.
//   Bw[(tap*64+c)*64 + n] = w_reg[n*576 + c*9 + tap]
//   Bo[(tap*64+c)*32 + n] = (n<18) ? w_off[(n*64+c)*9 + tap] : 0
// ---------------------------------------------------------------------------
__global__ __launch_bounds__(256)
void k_prep(const float* __restrict__ w_reg, const float* __restrict__ w_off,
            _Float16* __restrict__ Bw, _Float16* __restrict__ Bo) {
    const int stride = gridDim.x * blockDim.x;
    int tid = blockIdx.x * blockDim.x + threadIdx.x;
    for (int idx = tid; idx < KTOT * OC_; idx += stride) {
        const int n = idx & 63, k = idx >> 6;
        const int c = k & 63, tap = k >> 6;
        Bw[idx] = (_Float16)w_reg[n * KTOT + c * 9 + tap];
    }
    for (int idx = tid; idx < KTOT * 32; idx += stride) {
        const int n = idx & 31, k = idx >> 5;
        const int c = k & 63, tap = k >> 6;
        Bo[idx] = (n < OFFC) ? (_Float16)w_off[(n * 64 + c) * 9 + tap]
                             : (_Float16)0.f;
    }
}

// ---------------------------------------------------------------------------
// Kernel 1: offset conv (implicit GEMM, M=64px tile, N=32 (18 used), K=576).
// Grid: (W/64, H, B), 256 threads = 8 wave32, one 16x16 tile per wave.
// ---------------------------------------------------------------------------
__global__ __launch_bounds__(256)
void k_offsets(const float* __restrict__ x, const float* __restrict__ b_off,
               const _Float16* __restrict__ Bo, float* __restrict__ offs) {
    __shared__ _Float16 Xs[3 * 66 * A_LD];   // halo rows: [r][xx][c], 28.5 KB
    __shared__ float    Ds[XTILE * 32];      // 8 KB

    const int x0 = blockIdx.x * XTILE, y = blockIdx.y, b = blockIdx.z;
    const int tid = threadIdx.x;
    const float* __restrict__ xb = x + (size_t)b * C_ * HW_;

    // stage halo: x[b, c, y-1..y+1, x0-1..x0+64] -> f16, zero padded
    for (int idx = tid; idx < 3 * 66 * 64; idx += 256) {
        const int xx = idx % 66;
        const int rc = idx / 66;
        const int r = rc % 3, c = rc / 3;
        const int gy = y + r - 1, gx = x0 + xx - 1;
        float v = 0.f;
        if ((unsigned)gy < (unsigned)H_ && (unsigned)gx < (unsigned)W_)
            v = xb[(size_t)c * HW_ + gy * W_ + gx];
        Xs[(r * 66 + xx) * A_LD + c] = (_Float16)v;
    }
    __syncthreads();

    const int wv = tid >> 5, lane = tid & 31;
    const int mtile = wv & 3, ntile = wv >> 2;     // 4 M-tiles x 2 N-tiles
    const int mrow = mtile * 16 + (lane & 15);
    const int half = lane >> 4;
    const int n0 = ntile * 16;

    v8f acc = {};
    for (int tap = 0; tap < 9; ++tap) {
        const int i = tap / 3, j = tap % 3;
        const _Float16* __restrict__ arow = Xs + (i * 66 + mrow + j) * A_LD;
#pragma unroll
        for (int ch = 0; ch < 2; ++ch) {
            const int cbase = ch * 32 + half * 8;
            v16h a = cat8(*(const v8h*)(arow + cbase),
                          *(const v8h*)(arow + cbase + 16));
            const int kb = tap * 64 + ch * 32 + half * 8;   // B: N = lane&15, K in A's order
            v16h bm;
#pragma unroll
            for (int s = 0; s < 16; ++s)
                bm[s] = Bo[(size_t)(kb + (s < 8 ? s : s + 8)) * 32 + n0 + (lane & 15)];
            acc = wmma_f16(a, bm, acc);
        }
    }

    // D layout: M = v + 8*half, N = lane&15
#pragma unroll
    for (int v = 0; v < 8; ++v)
        Ds[(mtile * 16 + v + half * 8) * 32 + n0 + (lane & 15)] = acc[v];
    __syncthreads();

    for (int idx = tid; idx < OFFC * XTILE; idx += 256) {
        const int n = idx >> 6, p = idx & 63;
        offs[(((size_t)b * OFFC + n) * H_ + y) * W_ + x0 + p] =
            Ds[p * 32 + n] + b_off[n];
    }
}

// ---------------------------------------------------------------------------
// Kernel 2: deformed bilinear sampling + regression GEMM.
// Grid: (W/64, H, B), 256 threads = 8 wave32.
// Per tap: sample 64px x 64ch tile to LDS (f16), then WMMA K=64 (2 halves).
// Each wave owns 1 M-tile x 2 N-tiles (two v8f accumulators). Sampled tiles
// are double-buffered: gathers for tap t+1 overlap the WMMA stream of tap t.
// ---------------------------------------------------------------------------
__global__ __launch_bounds__(256)
void k_main(const float* __restrict__ x, const float* __restrict__ b_reg,
            const float* __restrict__ offs, const _Float16* __restrict__ Bw,
            float* __restrict__ out) {
    __shared__ _Float16 samp[2][XTILE * A_LD]; // double-buffered tap tile, 18 KB
    __shared__ float    offL[OFFC * XTILE];    // offsets for the segment, 4.5 KB
    __shared__ float    Ds[XTILE * OC_];       // output staging, 16 KB

    const int x0 = blockIdx.x * XTILE, y = blockIdx.y, b = blockIdx.z;
    const int tid = threadIdx.x;
    const float* __restrict__ xb = x + (size_t)b * C_ * HW_;

    for (int idx = tid; idx < OFFC * XTILE; idx += 256) {
        const int n = idx >> 6, p = idx & 63;
        offL[idx] = offs[(((size_t)b * OFFC + n) * H_ + y) * W_ + x0 + p];
    }
    __syncthreads();

    const int wv = tid >> 5, lane = tid & 31;
    const int mtile = wv & 3, npair = wv >> 2;     // 4 M-tiles x (2 N-tile pairs)
    const int n0 = npair * 32;
    const int mrow = mtile * 16 + (lane & 15);
    const int half = lane >> 4;

    // sampling role: pixel p, 16 channels starting at cb
    const int p = tid & 63;
    const int cb = (tid >> 6) * 16;
    const float gx = (float)(x0 + p), gy = (float)y;
    const float scl = 129.f / 127.f;               // (Wp-1)/(W-1)

    // ---- bilinear sample of one tap into dst (fp32, branchless masks) ----
    auto sample_tap = [&](int tap, _Float16* __restrict__ dst) {
        const int i = tap / 3, j = tap % 3;
        const float fx = gx + (float)(j - 1) + offL[(2 * tap) * XTILE + p];
        const float fy = gy + (float)(i - 1) + offL[(2 * tap + 1) * XTILE + p];
        const float qx = fx * scl, qy = fy * scl;  // padded-image coords
        const float x0f = floorf(qx), y0f = floorf(qy);
        const float wx1 = qx - x0f, wy1 = qy - y0f;
        const float wx0 = 1.f - wx1, wy0 = 1.f - wy1;
        const int ix0 = (int)x0f, iy0 = (int)y0f;
        const int ix1 = ix0 + 1, iy1 = iy0 + 1;
        // value nonzero only for padded coords in [1, 128] (pad ring is zero)
        const float mx0 = (ix0 >= 1 && ix0 <= W_) ? 1.f : 0.f;
        const float mx1 = (ix1 >= 1 && ix1 <= W_) ? 1.f : 0.f;
        const float my0 = (iy0 >= 1 && iy0 <= H_) ? 1.f : 0.f;
        const float my1 = (iy1 >= 1 && iy1 <= H_) ? 1.f : 0.f;
        const float w00 = wy0 * wx0 * my0 * mx0, w01 = wy0 * wx1 * my0 * mx1;
        const float w10 = wy1 * wx0 * my1 * mx0, w11 = wy1 * wx1 * my1 * mx1;
        const int cx0 = min(max(ix0, 1), W_) - 1, cx1 = min(max(ix1, 1), W_) - 1;
        const int cy0 = min(max(iy0, 1), H_) - 1, cy1 = min(max(iy1, 1), H_) - 1;
        const int o00 = cy0 * W_ + cx0, o01 = cy0 * W_ + cx1;
        const int o10 = cy1 * W_ + cx0, o11 = cy1 * W_ + cx1;
#pragma unroll 4
        for (int cc = 0; cc < 16; ++cc) {
            const float* __restrict__ pc = xb + (size_t)(cb + cc) * HW_;
            const float s = w00 * pc[o00] + w01 * pc[o01] +
                            w10 * pc[o10] + w11 * pc[o11];
            dst[p * A_LD + cb + cc] = (_Float16)s;
        }
    };

    v8f acc0 = {}, acc1 = {};

    sample_tap(0, samp[0]);
    __syncthreads();

    for (int tap = 0; tap < 9; ++tap) {
        const _Float16* __restrict__ cur = samp[tap & 1];

        // issue next tap's gathers first so they overlap this tap's WMMAs
        if (tap < 8)
            sample_tap(tap + 1, samp[(tap + 1) & 1]);

        // prefetch the tap-after-next B rows into L2 (global_prefetch_b8)
        if (tap < 7)
            __builtin_prefetch(Bw + (size_t)((tap + 2) * 64 + lane) * OC_, 0, 3);

        // ---- WMMA: K = 64 for this tap (2 halves of 32) ----
        const _Float16* __restrict__ arow = cur + mrow * A_LD;
#pragma unroll
        for (int ch = 0; ch < 2; ++ch) {
            const int cbase = ch * 32 + half * 8;
            v16h a = cat8(*(const v8h*)(arow + cbase),
                          *(const v8h*)(arow + cbase + 16));
            const int kb = tap * 64 + ch * 32 + half * 8;   // B: N = lane&15, K in A's order
            v16h b0, b1;
#pragma unroll
            for (int s = 0; s < 16; ++s) {
                const _Float16* __restrict__ brow = Bw + (size_t)(kb + (s < 8 ? s : s + 8)) * OC_;
                b0[s] = brow[n0 + (lane & 15)];
                b1[s] = brow[n0 + 16 + (lane & 15)];
            }
            acc0 = wmma_f16(a, b0, acc0);
            acc1 = wmma_f16(a, b1, acc1);
        }
        __syncthreads();   // next-buffer writes visible; cur safe to reuse at tap+2
    }

    // ---- epilogue: stage D, add bias, coalesced NCHW store ----
#pragma unroll
    for (int v = 0; v < 8; ++v) {
        const int row = mtile * 16 + v + half * 8;
        Ds[row * OC_ + n0 + (lane & 15)]      = acc0[v];
        Ds[row * OC_ + n0 + 16 + (lane & 15)] = acc1[v];
    }
    __syncthreads();
    for (int idx = tid; idx < XTILE * OC_; idx += 256) {
        const int oc = idx >> 6, pp = idx & 63;
        out[(((size_t)b * OC_ + oc) * H_ + y) * W_ + x0 + pp] =
            Ds[pp * OC_ + oc] + b_reg[oc];
    }
}

// ---------------------------------------------------------------------------
extern "C" void kernel_launch(void* const* d_in, const int* in_sizes, int n_in,
                              void* d_out, int out_size, void* d_ws, size_t ws_size,
                              hipStream_t stream) {
    (void)in_sizes; (void)n_in; (void)out_size; (void)ws_size;
    const float* x     = (const float*)d_in[0];
    const float* w_off = (const float*)d_in[1];
    const float* b_off = (const float*)d_in[2];
    const float* w_reg = (const float*)d_in[3];
    const float* b_reg = (const float*)d_in[4];
    float* out = (float*)d_out;

    char* ws = (char*)d_ws;
    float*    offs = (float*)ws;
    _Float16* Bw   = (_Float16*)(ws + WS_BW_OFF);
    _Float16* Bo   = (_Float16*)(ws + WS_BO_OFF);

    k_prep<<<64, 256, 0, stream>>>(w_reg, w_off, Bw, Bo);

    dim3 grid(W_ / XTILE, H_, B_);
    k_offsets<<<grid, 256, 0, stream>>>(x, b_off, Bo, offs);
    k_main<<<grid, 256, 0, stream>>>(x, b_reg, offs, Bw, out);
}
